// GeneralizedConvAttender_58617713656244
// MI455X (gfx1250) — matrix operand order repaired and checked by reading, not compiled, not run on hardware
//
#include <hip/hip_runtime.h>
#include <hip/hip_bf16.h>
#include <math.h>

typedef __attribute__((ext_vector_type(16))) _Float16 v16h;
typedef __attribute__((ext_vector_type(8)))  float    v8f;

#define NB   16
#define NK_  2048
#define NQ_  2048
#define T_   1024
#define V_   64
#define C_   64

// ---------------- workspace layout (bytes) ----------------
static const size_t OFF_VALH  = 0;                              // [B][V][NK] f16 : 4 MiB
static const size_t OFF_GVRAW = (size_t)4  << 20;               // [B][T][V] f32  : 4 MiB
static const size_t OFF_DENS  = (size_t)8  << 20;               // [B][T]    f32  : 64 KiB
static const size_t OFF_GVLIN = ((size_t)8 << 20) + (1u << 16); // [B][T][C] f32  : 4 MiB
static const size_t OFF_GVT   = ((size_t)12 << 20) + (1u << 16);// [B][C][T] f16  : 2 MiB
static const size_t OFF_FEAT  = ((size_t)14 << 20) + (1u << 16);// [B][NQ][132] f32

__device__ __forceinline__ float softplus_f(float x) { return __logf(1.0f + __expf(x)); }
__device__ __forceinline__ float sigmoid_f(float x)  { return 1.0f / (1.0f + __expf(-x)); }

__device__ __forceinline__ v16h make_ones() {
    v16h o;
#pragma unroll
    for (int i = 0; i < 16; ++i) o[i] = (_Float16)1.0f;
    return o;
}

// CDNA5 async global->LDS copy (ASYNCcnt-tracked), 16B per lane
__device__ __forceinline__ void async_b128(void* ldsPtr, const void* gPtr) {
    unsigned ldsOff = (unsigned)(uintptr_t)ldsPtr;                 // addr[31:0] = LDS byte address
    unsigned long long g = (unsigned long long)(uintptr_t)gPtr;
    asm volatile("global_load_async_to_lds_b128 %0, %1, off" :: "v"(ldsOff), "v"(g) : "memory");
}
__device__ __forceinline__ void wait_async0() {
    asm volatile("s_wait_asynccnt 0x0" ::: "memory");
}

// stage one 64-row x 128-K f16 B-slice into LDS (rows: stride srcStride elems)
__device__ __forceinline__ void stage_slice(_Float16* dst, const _Float16* src,
                                            int srcStride, int tid) {
#pragma unroll
    for (int t = 0; t < 4; ++t) {
        int idx = tid + t * 256;          // 1024 chunks of 8 halfs
        int row = idx >> 4;
        int ch  = idx & 15;
        async_b128(dst + row * 128 + ch * 8, src + (size_t)row * srcStride + ch * 8);
    }
}

// generate two 16x32 f16 A-tiles (rows r0S, r1S) from pre-scaled key values in LDS
// w = exp2(-(kv' - r')^2)
__device__ __forceinline__ void agen2(const float* kLds, int kb, int hi,
                                      float r0S, float r1S, v16h& a0, v16h& a1) {
#pragma unroll
    for (int g = 0; g < 2; ++g) {
        const float4* kp = reinterpret_cast<const float4*>(&kLds[kb + hi * 16 + g * 8]);
        float4 k0 = kp[0], k1 = kp[1];
        float kvv[8] = {k0.x, k0.y, k0.z, k0.w, k1.x, k1.y, k1.z, k1.w};
#pragma unroll
        for (int jj = 0; jj < 8; ++jj) {
            float d0 = kvv[jj] - r0S;
            float d1 = kvv[jj] - r1S;
            a0[g * 8 + jj] = (_Float16)__builtin_amdgcn_exp2f(-(d0 * d0));
            a1[g * 8 + jj] = (_Float16)__builtin_amdgcn_exp2f(-(d1 * d1));
        }
    }
}

#define WMMA_F16(A, B, CACC) \
    __builtin_amdgcn_wmma_f32_16x16x32_f16(false, (A), false, (B), (short)0, (CACC), false, false)

// ---------------------------------------------------------------------------
// values [B][NK][V] f32 -> valh [B][V][NK] f16
// ---------------------------------------------------------------------------
__global__ __launch_bounds__(256) void k_pack(const float* __restrict__ values,
                                              _Float16* __restrict__ valh) {
    int idx = blockIdx.x * 256 + threadIdx.x;
    int v = idx & 63;
    int k = (idx >> 6) & 2047;
    int b = idx >> 17;
    valh[(size_t)b * (V_ * NK_) + (size_t)v * NK_ + k] = (_Float16)values[idx];
}

// ---------------------------------------------------------------------------
// Stage 1: keys -> grid.  Wave = two 16-row t-tiles x 64 v.  K over NK.
// Async-staged B slices in LDS, 5th WMMA vs ones gives L1-normalizer rows.
// ---------------------------------------------------------------------------
__global__ __launch_bounds__(256) void k_stage1(const float* __restrict__ keys,
                                                const _Float16* __restrict__ valh,
                                                const float* __restrict__ ls_p,
                                                const float* __restrict__ dens_w,
                                                const float* __restrict__ dens_b,
                                                float* __restrict__ gv_raw,
                                                float* __restrict__ dens_sig) {
    __shared__ float    kLds[NK_];
    __shared__ _Float16 bstage[2][64 * 128];
    const int tid  = threadIdx.x;
    const int wave = tid >> 5, lane = tid & 31;
    const int task = blockIdx.x * 8 + wave;           // 512 tasks (32 per b)
    const int b    = task >> 5;
    const int t0   = (task & 31) << 5;

    // async-stage keys[b] (8 KiB), then scale by S in place
#pragma unroll
    for (int t = 0; t < 2; ++t) {
        int idx = tid + t * 256;                       // 512 chunks of 4 floats
        async_b128(&kLds[idx * 4], keys + (size_t)b * NK_ + idx * 4);
    }
    const float s = softplus_f(ls_p[0]);
    const float S = sqrtf(1.4426950408889634f) / s;    // w = exp2(-((kv-r)*S)^2)
    wait_async0();
    __syncthreads();
    for (int i = tid; i < NK_; i += 256) kLds[i] *= S;
    __syncthreads();

    const int m = lane & 15, hi = lane >> 4;
    const float r0S = (-1.0f + (float)(t0 + m)      * (2.0f / 1023.0f)) * S;
    const float r1S = (-1.0f + (float)(t0 + 16 + m) * (2.0f / 1023.0f)) * S;
    const v16h ones = make_ones();
    v8f acc0[5] = {}, acc1[5] = {};
    const _Float16* vb = valh + ((size_t)b << 17);

    stage_slice(&bstage[0][0], vb, NK_, tid);
    for (int sl = 0; sl < 16; ++sl) {
        wait_async0();
        __syncthreads();
        if (sl + 1 < 16) stage_slice(&bstage[(sl + 1) & 1][0], vb + (sl + 1) * 128, NK_, tid);
        const _Float16* bl = &bstage[sl & 1][0];
#pragma unroll
        for (int st = 0; st < 4; ++st) {
            const int kb = sl * 128 + st * 32, kl = st * 32;
            v16h a0, a1;
            agen2(kLds, kb, hi, r0S, r1S, a0, a1);
#pragma unroll
            for (int nt = 0; nt < 4; ++nt) {
                const v16h bm = *reinterpret_cast<const v16h*>(bl + (nt * 16 + m) * 128 + kl + hi * 16);
                acc0[nt] = WMMA_F16(a0, bm, acc0[nt]);
                acc1[nt] = WMMA_F16(a1, bm, acc1[nt]);
            }
            acc0[4] = WMMA_F16(a0, ones, acc0[4]);
            acc1[4] = WMMA_F16(a1, ones, acc1[4]);
        }
    }

    const float dw0 = dens_w[0], db0 = dens_b[0];
#pragma unroll
    for (int r = 0; r < 8; ++r) {
        const int mr = r + 8 * hi;
        {
            float rsum = acc0[4][r];
            float invd = 1.0f / fmaxf(rsum, 1e-12f);
            size_t row = (size_t)(b * T_ + t0 + mr);
#pragma unroll
            for (int nt = 0; nt < 4; ++nt)
                gv_raw[row * V_ + nt * 16 + m] = acc0[nt][r] * invd;
            if (m == 0) dens_sig[row] = sigmoid_f(fmaf(-rsum, dw0, db0));
        }
        {
            float rsum = acc1[4][r];
            float invd = 1.0f / fmaxf(rsum, 1e-12f);
            size_t row = (size_t)(b * T_ + t0 + 16 + mr);
#pragma unroll
            for (int nt = 0; nt < 4; ++nt)
                gv_raw[row * V_ + nt * 16 + m] = acc1[nt][r] * invd;
            if (m == 0) dens_sig[row] = sigmoid_f(fmaf(-rsum, dw0, db0));
        }
    }
}

// ---------------------------------------------------------------------------
// Linear: gvlin[b,t,c] = lin_b[c] + dens*lin_w[64,c] + sum_v gv_raw*lin_w[v,c]
// ---------------------------------------------------------------------------
__global__ __launch_bounds__(256) void k_linear(const float* __restrict__ gv_raw,
                                                const float* __restrict__ dens_sig,
                                                const float* __restrict__ lin_w,
                                                const float* __restrict__ lin_b,
                                                float* __restrict__ gvlin) {
    int idx = blockIdx.x * 256 + threadIdx.x;
    int c   = idx & 63;
    int row = idx >> 6;
    float acc = fmaf(dens_sig[row], lin_w[64 * C_ + c], lin_b[c]);
    const float* g = gv_raw + (size_t)row * V_;
#pragma unroll 8
    for (int v = 0; v < V_; ++v) acc = fmaf(g[v], lin_w[v * C_ + c], acc);
    gvlin[idx] = acc;
}

// ---------------------------------------------------------------------------
// Conv stack on reshape-view x[b][c][t] = gvlin_flat[b][c*T + t]; 5 layers in LDS
// ---------------------------------------------------------------------------
__global__ __launch_bounds__(256) void k_conv(const float* __restrict__ gvlin,
                                              const float* __restrict__ dw_w,
                                              const float* __restrict__ dw_b,
                                              const float* __restrict__ pw_w,
                                              const float* __restrict__ pw_b,
                                              _Float16* __restrict__ gvT) {
    __shared__ float xs[64 * 125];
    __shared__ float ys[64 * 125];
    const int b     = blockIdx.x >> 4;
    const int chunk = blockIdx.x & 15;
    const int t0    = chunk * 64;
    const float* gf = gvlin + (size_t)b * (C_ * T_);

    for (int idx = threadIdx.x; idx < 64 * 124; idx += 256) {
        int c = idx / 124, i = idx - c * 124;
        int t = t0 - 30 + i;
        xs[c * 125 + i] = (t >= 0 && t < T_) ? gf[c * T_ + t] : 0.0f;
    }
    __syncthreads();

    const int c     = threadIdx.x & 63;
    const int slice = threadIdx.x >> 6;

    for (int layer = 0; layer < 5; ++layer) {
        const int lo = 6 * (layer + 1);
        const int hiB = 124 - 6 * (layer + 1);

        float wdw[7];
#pragma unroll
        for (int j = 0; j < 7; ++j) wdw[j] = dw_w[(layer * C_ + c) * 7 + j];
        const float bdw = dw_b[layer * C_ + c];

        for (int i = lo + slice; i < hiB; i += 4) {
            int t = t0 - 30 + i;
            if (t >= 0 && t < T_) {
                float acc = bdw;
#pragma unroll
                for (int j = 0; j < 7; ++j)
                    acc = fmaf(wdw[j], fmaxf(xs[c * 125 + i + 2 * j - 6], 0.0f), acc);
                ys[c * 125 + i] = acc;
            }
        }
        __syncthreads();

        float wpw[64];
#pragma unroll
        for (int cc = 0; cc < 64; ++cc) wpw[cc] = pw_w[(layer * C_ + c) * C_ + cc];
        const float bpw = pw_b[layer * C_ + c];

        for (int i = lo + slice; i < hiB; i += 4) {
            int t = t0 - 30 + i;
            if (t >= 0 && t < T_) {
                float acc = bpw;
#pragma unroll
                for (int cc = 0; cc < 64; ++cc) acc = fmaf(wpw[cc], ys[cc * 125 + i], acc);
                xs[c * 125 + i] += acc;
            }
        }
        __syncthreads();
    }

    for (int idx = threadIdx.x; idx < 64 * 64; idx += 256) {
        int cc = idx >> 6;
        int it = idx & 63;
        int t  = t0 + it;
        float vvv = xs[cc * 125 + 30 + it];
        gvT[(size_t)b * (C_ * T_) + (size_t)(t & 63) * T_ + (cc * 16 + chunk)] = (_Float16)vvv;
    }
}

// ---------------------------------------------------------------------------
// Stage 2a: grid -> queries (analytic tq).  feat cols [0,64)
// ---------------------------------------------------------------------------
__global__ __launch_bounds__(256) void k_stage2q(const float* __restrict__ queries,
                                                 const _Float16* __restrict__ gvT,
                                                 const float* __restrict__ ls_p,
                                                 float* __restrict__ feat) {
    __shared__ _Float16 bstage[2][64 * 128];
    const int tid  = threadIdx.x;
    const int wave = tid >> 5, lane = tid & 31;
    const int task = blockIdx.x * 8 + wave;           // 1024 tasks (64 per b)
    const int b    = task >> 6;
    const int q0   = (task & 63) << 5;

    const float s = softplus_f(ls_p[0]);
    const float S = sqrtf(1.4426950408889634f) / s;
    const int m = lane & 15, hi = lane >> 4;
    const float q0S = queries[(size_t)b * NQ_ + q0 + m] * S;
    const float q1S = queries[(size_t)b * NQ_ + q0 + 16 + m] * S;
    const float stepS = (2.0f / 1023.0f) * S;

    float tb[16];
#pragma unroll
    for (int g = 0; g < 2; ++g)
#pragma unroll
        for (int jj = 0; jj < 8; ++jj)
            tb[g * 8 + jj] = (-1.0f + (float)(hi * 16 + g * 8 + jj) * (2.0f / 1023.0f)) * S;

    const v16h ones = make_ones();
    v8f acc0[5] = {}, acc1[5] = {};
    const _Float16* gb = gvT + (size_t)b * (C_ * T_);

    stage_slice(&bstage[0][0], gb, T_, tid);
    for (int sl = 0; sl < 8; ++sl) {
        wait_async0();
        __syncthreads();
        if (sl + 1 < 8) stage_slice(&bstage[(sl + 1) & 1][0], gb + (sl + 1) * 128, T_, tid);
        const _Float16* bl = &bstage[sl & 1][0];
#pragma unroll
        for (int st = 0; st < 4; ++st) {
            const int kb = sl * 128 + st * 32, kl = st * 32;
            const float u0 = q0S - (float)kb * stepS;
            const float u1 = q1S - (float)kb * stepS;
            v16h a0, a1;
#pragma unroll
            for (int e = 0; e < 16; ++e) {
                float d0 = u0 - tb[e];
                float d1 = u1 - tb[e];
                a0[e] = (_Float16)__builtin_amdgcn_exp2f(-(d0 * d0));
                a1[e] = (_Float16)__builtin_amdgcn_exp2f(-(d1 * d1));
            }
#pragma unroll
            for (int nt = 0; nt < 4; ++nt) {
                const v16h bm = *reinterpret_cast<const v16h*>(bl + (nt * 16 + m) * 128 + kl + hi * 16);
                acc0[nt] = WMMA_F16(a0, bm, acc0[nt]);
                acc1[nt] = WMMA_F16(a1, bm, acc1[nt]);
            }
            acc0[4] = WMMA_F16(a0, ones, acc0[4]);
            acc1[4] = WMMA_F16(a1, ones, acc1[4]);
        }
    }

#pragma unroll
    for (int r = 0; r < 8; ++r) {
        const int mr = r + 8 * hi;
        {
            float invd = 1.0f / fmaxf(acc0[4][r], 1e-12f);
            size_t row = (size_t)(b * NQ_ + q0 + mr);
#pragma unroll
            for (int nt = 0; nt < 4; ++nt)
                feat[row * 132 + nt * 16 + m] = acc0[nt][r] * invd;
        }
        {
            float invd = 1.0f / fmaxf(acc1[4][r], 1e-12f);
            size_t row = (size_t)(b * NQ_ + q0 + 16 + mr);
#pragma unroll
            for (int nt = 0; nt < 4; ++nt)
                feat[row * 132 + nt * 16 + m] = acc1[nt][r] * invd;
        }
    }
}

// ---------------------------------------------------------------------------
// Stage 2b: direct keys -> queries + density.  feat cols [64,128) and 128
// ---------------------------------------------------------------------------
__global__ __launch_bounds__(256) void k_stage2k(const float* __restrict__ queries,
                                                 const float* __restrict__ keys,
                                                 const _Float16* __restrict__ valh,
                                                 const float* __restrict__ ls_p,
                                                 const float* __restrict__ dens_w,
                                                 const float* __restrict__ dens_b,
                                                 float* __restrict__ feat) {
    __shared__ float    kLds[NK_];
    __shared__ _Float16 bstage[2][64 * 128];
    const int tid  = threadIdx.x;
    const int wave = tid >> 5, lane = tid & 31;
    const int task = blockIdx.x * 8 + wave;           // 1024 tasks (64 per b)
    const int b    = task >> 6;
    const int q0   = (task & 63) << 5;

#pragma unroll
    for (int t = 0; t < 2; ++t) {
        int idx = tid + t * 256;
        async_b128(&kLds[idx * 4], keys + (size_t)b * NK_ + idx * 4);
    }
    const float s = softplus_f(ls_p[0]);
    const float S = sqrtf(1.4426950408889634f) / s;
    wait_async0();
    __syncthreads();
    for (int i = tid; i < NK_; i += 256) kLds[i] *= S;
    __syncthreads();

    const int m = lane & 15, hi = lane >> 4;
    const float r0S = queries[(size_t)b * NQ_ + q0 + m] * S;
    const float r1S = queries[(size_t)b * NQ_ + q0 + 16 + m] * S;
    const v16h ones = make_ones();
    v8f acc0[5] = {}, acc1[5] = {};
    const _Float16* vb = valh + ((size_t)b << 17);

    stage_slice(&bstage[0][0], vb, NK_, tid);
    for (int sl = 0; sl < 16; ++sl) {
        wait_async0();
        __syncthreads();
        if (sl + 1 < 16) stage_slice(&bstage[(sl + 1) & 1][0], vb + (sl + 1) * 128, NK_, tid);
        const _Float16* bl = &bstage[sl & 1][0];
#pragma unroll
        for (int st = 0; st < 4; ++st) {
            const int kb = sl * 128 + st * 32, kl = st * 32;
            v16h a0, a1;
            agen2(kLds, kb, hi, r0S, r1S, a0, a1);
#pragma unroll
            for (int nt = 0; nt < 4; ++nt) {
                const v16h bm = *reinterpret_cast<const v16h*>(bl + (nt * 16 + m) * 128 + kl + hi * 16);
                acc0[nt] = WMMA_F16(a0, bm, acc0[nt]);
                acc1[nt] = WMMA_F16(a1, bm, acc1[nt]);
            }
            acc0[4] = WMMA_F16(a0, ones, acc0[4]);
            acc1[4] = WMMA_F16(a1, ones, acc1[4]);
        }
    }

    const float dw0 = dens_w[0], db0 = dens_b[0];
#pragma unroll
    for (int r = 0; r < 8; ++r) {
        const int mr = r + 8 * hi;
        {
            float rsum = acc0[4][r];
            float invd = 1.0f / fmaxf(rsum, 1e-12f);
            size_t row = (size_t)(b * NQ_ + q0 + mr);
#pragma unroll
            for (int nt = 0; nt < 4; ++nt)
                feat[row * 132 + 64 + nt * 16 + m] = acc0[nt][r] * invd;
            if (m == 0) feat[row * 132 + 128] = sigmoid_f(fmaf(-rsum, dw0, db0));
        }
        {
            float rsum = acc1[4][r];
            float invd = 1.0f / fmaxf(rsum, 1e-12f);
            size_t row = (size_t)(b * NQ_ + q0 + 16 + mr);
#pragma unroll
            for (int nt = 0; nt < 4; ++nt)
                feat[row * 132 + 64 + nt * 16 + m] = acc1[nt][r] * invd;
            if (m == 0) feat[row * 132 + 128] = sigmoid_f(fmaf(-rsum, dw0, db0));
        }
    }
}

// ---------------------------------------------------------------------------
// MLP head: 129 -> 16 -> 16 -> 16 -> 128
// ---------------------------------------------------------------------------
__global__ __launch_bounds__(256) void k_mlp(const float* __restrict__ feat,
                                             const float* __restrict__ w0, const float* __restrict__ b0,
                                             const float* __restrict__ w1, const float* __restrict__ b1,
                                             const float* __restrict__ w2, const float* __restrict__ b2,
                                             const float* __restrict__ w3, const float* __restrict__ b3,
                                             float* __restrict__ out) {
    __shared__ float sw0[129 * 16], sw1[256], sw2[256], sw3[16 * 128];
    __shared__ float sb0[16], sb1[16], sb2[16], sb3[128];
    for (int i = threadIdx.x; i < 129 * 16; i += 256) sw0[i] = w0[i];
    for (int i = threadIdx.x; i < 256; i += 256) { sw1[i] = w1[i]; sw2[i] = w2[i]; }
    for (int i = threadIdx.x; i < 16 * 128; i += 256) sw3[i] = w3[i];
    if (threadIdx.x < 16) {
        sb0[threadIdx.x] = b0[threadIdx.x];
        sb1[threadIdx.x] = b1[threadIdx.x];
        sb2[threadIdx.x] = b2[threadIdx.x];
    }
    if (threadIdx.x < 128) sb3[threadIdx.x] = b3[threadIdx.x];
    __syncthreads();

    const int q = blockIdx.x * 256 + threadIdx.x;
    const float* f = feat + (size_t)q * 132;

    float h0[16], h1[16], h2[16];
#pragma unroll
    for (int o = 0; o < 16; ++o) h0[o] = sb0[o];
    for (int j = 0; j < 129; ++j) {
        float fv = f[j];
#pragma unroll
        for (int o = 0; o < 16; ++o) h0[o] = fmaf(fv, sw0[j * 16 + o], h0[o]);
    }
#pragma unroll
    for (int o = 0; o < 16; ++o) { h0[o] = fmaxf(h0[o], 0.0f); h1[o] = sb1[o]; }
#pragma unroll
    for (int j = 0; j < 16; ++j) {
        float fv = h0[j];
#pragma unroll
        for (int o = 0; o < 16; ++o) h1[o] = fmaf(fv, sw1[j * 16 + o], h1[o]);
    }
#pragma unroll
    for (int o = 0; o < 16; ++o) { h1[o] = fmaxf(h1[o], 0.0f); h2[o] = sb2[o]; }
#pragma unroll
    for (int j = 0; j < 16; ++j) {
        float fv = h1[j];
#pragma unroll
        for (int o = 0; o < 16; ++o) h2[o] = fmaf(fv, sw2[j * 16 + o], h2[o]);
    }
#pragma unroll
    for (int o = 0; o < 16; ++o) h2[o] = fmaxf(h2[o], 0.0f);

    float* op = out + (size_t)q * 128;
    for (int o2 = 0; o2 < 128; ++o2) {
        float acc = sb3[o2];
#pragma unroll
        for (int j = 0; j < 16; ++j) acc = fmaf(h2[j], sw3[j * 128 + o2], acc);
        op[o2] = acc;
    }
}

// ---------------------------------------------------------------------------
extern "C" void kernel_launch(void* const* d_in, const int* in_sizes, int n_in,
                              void* d_out, int out_size, void* d_ws, size_t ws_size,
                              hipStream_t stream) {
    const float* keys    = (const float*)d_in[0];
    const float* queries = (const float*)d_in[1];
    const float* values  = (const float*)d_in[2];
    const float* ls      = (const float*)d_in[3];
    const float* lin_w   = (const float*)d_in[4];
    const float* lin_b   = (const float*)d_in[5];
    const float* dens_w  = (const float*)d_in[6];
    const float* dens_b  = (const float*)d_in[7];
    const float* dw_w    = (const float*)d_in[8];
    const float* dw_b    = (const float*)d_in[9];
    const float* pw_w    = (const float*)d_in[10];
    const float* pw_b    = (const float*)d_in[11];
    const float* w0      = (const float*)d_in[12];
    const float* b0      = (const float*)d_in[13];
    const float* w1      = (const float*)d_in[14];
    const float* b1      = (const float*)d_in[15];
    const float* w2      = (const float*)d_in[16];
    const float* b2      = (const float*)d_in[17];
    const float* w3      = (const float*)d_in[18];
    const float* b3      = (const float*)d_in[19];

    char* ws = (char*)d_ws;
    _Float16* valh   = (_Float16*)(ws + OFF_VALH);
    float*    gv_raw = (float*)   (ws + OFF_GVRAW);
    float*    dens1  = (float*)   (ws + OFF_DENS);
    float*    gvlin  = (float*)   (ws + OFF_GVLIN);
    _Float16* gvT    = (_Float16*)(ws + OFF_GVT);
    float*    feat   = (float*)   (ws + OFF_FEAT);

    k_pack   <<<8192, 256, 0, stream>>>(values, valh);
    k_stage1 <<<64,   256, 0, stream>>>(keys, valh, ls, dens_w, dens_b, gv_raw, dens1);
    k_linear <<<4096, 256, 0, stream>>>(gv_raw, dens1, lin_w, lin_b, gvlin);
    k_conv   <<<256,  256, 0, stream>>>(gvlin, dw_w, dw_b, pw_w, pw_b, gvT);
    k_stage2q<<<128,  256, 0, stream>>>(queries, gvT, ls, feat);
    k_stage2k<<<128,  256, 0, stream>>>(queries, keys, valh, ls, dens_w, dens_b, feat);
    k_mlp    <<<128,  256, 0, stream>>>(feat, w0, b0, w1, b1, w2, b2, w3, b3, (float*)d_out);
}
